// GATModel_37598143709433
// MI455X (gfx1250) — matrix operand order repaired and checked by reading, not compiled, not run on hardware
//
#include <hip/hip_runtime.h>
#include <hip/hip_bf16.h>
#include <math.h>

// ---------------------------------------------------------------------------
// GAT (2-layer, PyG-style) for MI455X / gfx1250.
//   N=100000 nodes, E=1.6M edges (+N self loops), HID=64, HEADS=4, G=64.
// Memory-bound (~5 GB traffic -> ~215us @ 23.3 TB/s). The one real GEMM
// (h[100k,256] @ W2[256,64], 3.3 GFLOP) goes through v_wmma_f32_16x16x32_bf16.
// ---------------------------------------------------------------------------

typedef __bf16 bf16;
typedef bf16  v16bf __attribute__((ext_vector_type(16)));
typedef bf16  v8bf  __attribute__((ext_vector_type(8)));
typedef float v8f   __attribute__((ext_vector_type(8)));

#define HEADS 4
#define HID   64
#define F1    256   // HEADS*HID
#define G     64
#define TASKS 2
#define NEG_SLOPE 0.2f

// Monotonic float -> uint key (total order) so we can use integer atomicMax.
__device__ __forceinline__ unsigned fkey(float f) {
    unsigned u = __float_as_uint(f);
    return (u & 0x80000000u) ? ~u : (u | 0x80000000u);
}
__device__ __forceinline__ float funkey(unsigned k) {
    unsigned u = (k & 0x80000000u) ? (k & 0x7fffffffu) : ~k;
    return __uint_as_float(u);
}
__device__ __forceinline__ float lrelu(float v) { return v > 0.f ? v : NEG_SLOPE * v; }
__device__ __forceinline__ float elu(float v)   { return v > 0.f ? v : expm1f(v); }

// ---------------------------------------------------------------------------
__global__ void k_zero(unsigned* __restrict__ p, long n) {
    long i = blockIdx.x * (long)blockDim.x + threadIdx.x;
    long s = gridDim.x * (long)blockDim.x;
    for (; i < n; i += s) p[i] = 0u;
}

// xh1 = x @ W1 (K=2), plus per-head attention logits a_src1/a_dst1.
// One block per node, 256 threads = one feature channel each.
__global__ void k_xh1(const float* __restrict__ x, const float* __restrict__ W1,
                      const float* __restrict__ as1, const float* __restrict__ ad1,
                      float* __restrict__ xh1, float* __restrict__ a_src,
                      float* __restrict__ a_dst) {
    __shared__ float ss[F1];
    __shared__ float sd[F1];
    int n = blockIdx.x, j = threadIdx.x;
    float x0 = x[n * 2 + 0], x1 = x[n * 2 + 1];
    float v = fmaf(x0, W1[j], x1 * W1[F1 + j]);
    xh1[(size_t)n * F1 + j] = v;
    ss[j] = v * as1[j];
    sd[j] = v * ad1[j];
    __syncthreads();
    #pragma unroll
    for (int s = 32; s >= 1; s >>= 1) {
        if ((j & 63) < s) { ss[j] += ss[j + s]; sd[j] += sd[j + s]; }
        __syncthreads();
    }
    if ((j & 63) == 0) {
        a_src[n * HEADS + (j >> 6)] = ss[j];
        a_dst[n * HEADS + (j >> 6)] = sd[j];
    }
}

__global__ void k_w2bf(const float* __restrict__ W2, bf16* __restrict__ w2bf) {
    int i = blockIdx.x * blockDim.x + threadIdx.x;
    if (i < F1 * HID) w2bf[i] = (bf16)W2[i];
}

// ------------------------- layer-1 edge softmax ----------------------------
__global__ void k_edge_max1(const int* __restrict__ ei, const float* __restrict__ asrc,
                            const float* __restrict__ adst, unsigned* __restrict__ emax,
                            int E, int N) {
    int e = blockIdx.x * blockDim.x + threadIdx.x;
    if (e >= E + N) return;
    int s = (e < E) ? ei[e] : (e - E);
    int d = (e < E) ? ei[E + e] : (e - E);
    #pragma unroll
    for (int h = 0; h < HEADS; ++h) {
        float v = lrelu(asrc[s * HEADS + h] + adst[d * HEADS + h]);
        atomicMax(&emax[d * HEADS + h], fkey(v));
    }
}

__global__ void k_edge_sum1(const int* __restrict__ ei, const float* __restrict__ asrc,
                            const float* __restrict__ adst, const unsigned* __restrict__ emax,
                            float* __restrict__ den, int E, int N) {
    int e = blockIdx.x * blockDim.x + threadIdx.x;
    if (e >= E + N) return;
    int s = (e < E) ? ei[e] : (e - E);
    int d = (e < E) ? ei[E + e] : (e - E);
    #pragma unroll
    for (int h = 0; h < HEADS; ++h) {
        float v = lrelu(asrc[s * HEADS + h] + adst[d * HEADS + h]);
        atomicAdd(&den[d * HEADS + h], __expf(v - funkey(emax[d * HEADS + h])));
    }
}

// 64 threads per edge, 4 channels each (float4 gather, 4 float atomics scatter).
__global__ void k_edge_msg1(const int* __restrict__ ei, const float* __restrict__ asrc,
                            const float* __restrict__ adst, const unsigned* __restrict__ emax,
                            const float* __restrict__ den, const float* __restrict__ xh1,
                            float* __restrict__ out1, int E, int N) {
    long t = blockIdx.x * (long)blockDim.x + threadIdx.x;
    int e = (int)(t >> 6);
    int j = (int)(t & 63);
    if (e >= E + N) return;
    int s = (e < E) ? ei[e] : (e - E);
    int d = (e < E) ? ei[E + e] : (e - E);
    int h = j >> 4;                               // channel base j*4 -> head
    __builtin_prefetch(xh1 + (size_t)s * F1 + j * 4, 0, 1);
    float v = lrelu(asrc[s * HEADS + h] + adst[d * HEADS + h]);
    float alpha = __expf(v - funkey(emax[d * HEADS + h])) / den[d * HEADS + h];
    float4 xv = *(const float4*)(xh1 + (size_t)s * F1 + j * 4);
    float* o = out1 + (size_t)d * F1 + j * 4;
    atomicAdd(o + 0, xv.x * alpha);
    atomicAdd(o + 1, xv.y * alpha);
    atomicAdd(o + 2, xv.z * alpha);
    atomicAdd(o + 3, xv.w * alpha);
}

// h = elu(out1 + b1) -> bf16 (feeds WMMA GEMM; halves A-matrix traffic).
__global__ void k_bias_elu1(const float* __restrict__ out1, const float* __restrict__ b1,
                            bf16* __restrict__ hbf, long total) {
    long t = blockIdx.x * (long)blockDim.x + threadIdx.x;
    if (t >= total) return;
    int j = (int)(t & (F1 - 1));
    hbf[t] = (bf16)elu(out1[t] + b1[j]);
}

// ------------------------- layer-2 GEMM via WMMA ---------------------------
// xh2[N,64] = h_bf16[N,256] @ W2_bf16[256,64]; K=256 -> 8 chained
// v_wmma_f32_16x16x32_bf16 per 16x16 tile. 8 waves/block: 2 M-tiles x 4 N-tiles.
// W2 staged transposed in LDS (32 KB) so B fragments are contiguous ds loads.
__global__ void k_gemm2(const bf16* __restrict__ hbf, const bf16* __restrict__ w2bf,
                        float* __restrict__ xh2, int N) {
    __shared__ bf16 ldsWt[HID * F1];              // [col][k], 32 KB
    int tid = threadIdx.x;
    for (int i = tid; i < HID * F1; i += 256) {
        int c = i >> 8, k = i & (F1 - 1);
        ldsWt[i] = w2bf[k * HID + c];
    }
    __syncthreads();

    int wave = tid >> 5, lane = tid & 31;
    int m0 = (blockIdx.x * 2 + (wave >> 2)) * 16;
    if (m0 >= N) return;
    int c  = (wave & 3) * 16 + (lane & 15);       // output column
    int hi = (lane >= 16);                        // lane half selects K sub-block
    int r  = m0 + (lane & 15);
    int rr = r < N ? r : N - 1;                   // clamp (N%16==0 in practice)
    const bf16* arow = hbf + (size_t)rr * F1;
    const bf16* bcol = ldsWt + c * F1;

    v8f acc = {0.f, 0.f, 0.f, 0.f, 0.f, 0.f, 0.f, 0.f};
    #pragma unroll
    for (int kb = 0; kb < F1; kb += 32) {
        // A 16x32 bf16 lane layout (ISA 7.12.2): lanes<16 hold K {0..7,16..23},
        // lanes>=16 hold K {8..15,24..31} of this 32-wide K slab.
        v8bf alo = *(const v8bf*)(arow + kb + (hi ? 8 : 0));
        v8bf ahi = *(const v8bf*)(arow + kb + 16 + (hi ? 8 : 0));
        v16bf a;
        #pragma unroll
        for (int i = 0; i < 8; ++i) { a[i] = alo[i]; a[8 + i] = ahi[i]; }
        // B 32x16: lanes<16 hold K {0..7,16..23}, lanes>=16 hold K {8..15,24..31} of the slab.
        v8bf b0 = *(const v8bf*)(bcol + kb + (hi ? 8 : 0));
        v8bf b1 = *(const v8bf*)(bcol + kb + 16 + (hi ? 8 : 0));
        v16bf b;
        #pragma unroll
        for (int i = 0; i < 8; ++i) { b[i] = b0[i]; b[8 + i] = b1[i]; }
        acc = __builtin_amdgcn_wmma_f32_16x16x32_bf16(
            /*neg_a=*/false, a, /*neg_b=*/false, b,
            /*c_mod=*/(short)0, acc, /*reuse_a=*/false, /*reuse_b=*/false);
    }
    // C/D layout: VGPR v holds (M = m0 + v + 8*hi, N = c).
    #pragma unroll
    for (int v = 0; v < 8; ++v) {
        int m = m0 + v + (hi ? 8 : 0);
        if (m < N) xh2[(size_t)m * HID + c] = acc[v];
    }
}

// Layer-2 attention logits: one wave per node, shuffle reduce over 64 channels.
__global__ void k_attn2(const float* __restrict__ xh2, const float* __restrict__ as2,
                        const float* __restrict__ ad2, float* __restrict__ a_src,
                        float* __restrict__ a_dst, int N) {
    int gt = blockIdx.x * blockDim.x + threadIdx.x;
    int n = gt >> 5, lane = gt & 31;
    if (n >= N) return;
    float v0 = xh2[(size_t)n * HID + lane];
    float v1 = xh2[(size_t)n * HID + 32 + lane];
    float s = v0 * as2[lane] + v1 * as2[32 + lane];
    float d = v0 * ad2[lane] + v1 * ad2[32 + lane];
    #pragma unroll
    for (int off = 16; off >= 1; off >>= 1) {
        s += __shfl_down(s, off, 32);
        d += __shfl_down(d, off, 32);
    }
    if (lane == 0) { a_src[n] = s; a_dst[n] = d; }
}

// ------------------------- layer-2 edge softmax ----------------------------
__global__ void k_edge_max2(const int* __restrict__ ei, const float* __restrict__ asrc,
                            const float* __restrict__ adst, unsigned* __restrict__ emax,
                            int E, int N) {
    int e = blockIdx.x * blockDim.x + threadIdx.x;
    if (e >= E + N) return;
    int s = (e < E) ? ei[e] : (e - E);
    int d = (e < E) ? ei[E + e] : (e - E);
    atomicMax(&emax[d], fkey(lrelu(asrc[s] + adst[d])));
}

__global__ void k_edge_sum2(const int* __restrict__ ei, const float* __restrict__ asrc,
                            const float* __restrict__ adst, const unsigned* __restrict__ emax,
                            float* __restrict__ den, int E, int N) {
    int e = blockIdx.x * blockDim.x + threadIdx.x;
    if (e >= E + N) return;
    int s = (e < E) ? ei[e] : (e - E);
    int d = (e < E) ? ei[E + e] : (e - E);
    float v = lrelu(asrc[s] + adst[d]);
    atomicAdd(&den[d], __expf(v - funkey(emax[d])));
}

// 16 threads per edge, float4 gather + 4 float atomics.
__global__ void k_edge_msg2(const int* __restrict__ ei, const float* __restrict__ asrc,
                            const float* __restrict__ adst, const unsigned* __restrict__ emax,
                            const float* __restrict__ den, const float* __restrict__ xh2,
                            float* __restrict__ out2, int E, int N) {
    long t = blockIdx.x * (long)blockDim.x + threadIdx.x;
    int e = (int)(t >> 4);
    int j = (int)(t & 15);
    if (e >= E + N) return;
    int s = (e < E) ? ei[e] : (e - E);
    int d = (e < E) ? ei[E + e] : (e - E);
    float v = lrelu(asrc[s] + adst[d]);
    float alpha = __expf(v - funkey(emax[d])) / den[d];
    float4 xv = *(const float4*)(xh2 + (size_t)s * HID + j * 4);
    float* o = out2 + (size_t)d * HID + j * 4;
    atomicAdd(o + 0, xv.x * alpha);
    atomicAdd(o + 1, xv.y * alpha);
    atomicAdd(o + 2, xv.z * alpha);
    atomicAdd(o + 3, xv.w * alpha);
}

// h2 = elu(out2 + b2); mean-pool accumulation into pooled[G,64] and counts[G].
__global__ void k_pool(const float* __restrict__ out2, const float* __restrict__ b2,
                       const int* __restrict__ batch, float* __restrict__ pooled,
                       float* __restrict__ cnts, int N) {
    long t = blockIdx.x * (long)blockDim.x + threadIdx.x;
    if (t >= (long)N * HID) return;
    int n = (int)(t >> 6), c = (int)(t & 63);
    float v = elu(out2[t] + b2[c]);
    int g = batch[n];
    atomicAdd(&pooled[g * HID + c], v);
    if (c == 0) atomicAdd(&cnts[g], 1.0f);
}

__global__ void k_final(const float* __restrict__ pooled, const float* __restrict__ cnts,
                        const float* __restrict__ Wl, const float* __restrict__ bl,
                        float* __restrict__ out) {
    int tid = threadIdx.x;
    if (tid >= G * TASKS) return;
    int g = tid >> 1, t = tid & 1;
    float inv = 1.0f / fmaxf(cnts[g], 1.0f);
    float acc = bl[t];
    #pragma unroll
    for (int c = 0; c < HID; ++c)
        acc = fmaf(pooled[g * HID + c] * inv, Wl[c * TASKS + t], acc);
    out[g * TASKS + t] = acc;
}

// ---------------------------------------------------------------------------
extern "C" void kernel_launch(void* const* d_in, const int* in_sizes, int n_in,
                              void* d_out, int out_size, void* d_ws, size_t ws_size,
                              hipStream_t stream) {
    const float* x   = (const float*)d_in[0];
    const int*   ei  = (const int*)  d_in[1];
    const int*   bat = (const int*)  d_in[2];
    const float* W1  = (const float*)d_in[3];
    const float* as1 = (const float*)d_in[4];
    const float* ad1 = (const float*)d_in[5];
    const float* b1  = (const float*)d_in[6];
    const float* W2  = (const float*)d_in[7];
    const float* as2 = (const float*)d_in[8];
    const float* ad2 = (const float*)d_in[9];
    const float* b2  = (const float*)d_in[10];
    const float* Wl  = (const float*)d_in[11];
    const float* bl  = (const float*)d_in[12];

    const int N  = in_sizes[0] / 2;   // x is [N,2]
    const int E  = in_sizes[1] / 2;   // edge_index is [2,E]
    const int Et = E + N;             // + self loops

    char* ws = (char*)d_ws;
    size_t off = 0;
    auto alloc = [&](size_t bytes) -> void* {
        void* p = ws + off;
        off = (off + bytes + 255) & ~(size_t)255;
        return p;
    };
    float*    xh1   = (float*)   alloc((size_t)N * F1 * 4);   // also reused as out2
    bf16*     hbf   = (bf16*)    alloc((size_t)N * F1 * 2);
    float*    asc1  = (float*)   alloc((size_t)N * HEADS * 4);
    float*    adt1  = (float*)   alloc((size_t)N * HEADS * 4);
    unsigned* emax1 = (unsigned*)alloc((size_t)N * HEADS * 4);
    float*    den1  = (float*)   alloc((size_t)N * HEADS * 4);
    float*    out1  = (float*)   alloc((size_t)N * F1 * 4);
    float*    xh2   = (float*)   alloc((size_t)N * HID * 4);
    float*    asc2  = (float*)   alloc((size_t)N * 4);
    float*    adt2  = (float*)   alloc((size_t)N * 4);
    unsigned* emax2 = (unsigned*)alloc((size_t)N * 4);
    float*    den2  = (float*)   alloc((size_t)N * 4);
    bf16*     w2bf  = (bf16*)    alloc((size_t)F1 * HID * 2);
    float*    pool  = (float*)   alloc((size_t)G * HID * 4);
    float*    cnts  = (float*)   alloc((size_t)G * 4);
    float*    out2  = xh1;  // xh1 is dead after k_edge_msg1

    const int T = 256;
    auto gr = [](long n, int t) { return (unsigned)((n + t - 1) / t); };

    // ---- layer 1 ----
    k_zero<<<gr((long)N * F1, T), T, 0, stream>>>((unsigned*)out1, (long)N * F1);
    k_zero<<<gr((long)N * HEADS, T), T, 0, stream>>>(emax1, (long)N * HEADS);
    k_zero<<<gr((long)N * HEADS, T), T, 0, stream>>>((unsigned*)den1, (long)N * HEADS);

    k_xh1<<<N, F1, 0, stream>>>(x, W1, as1, ad1, xh1, asc1, adt1);
    k_w2bf<<<gr(F1 * HID, T), T, 0, stream>>>(W2, w2bf);

    k_edge_max1<<<gr(Et, T), T, 0, stream>>>(ei, asc1, adt1, emax1, E, N);
    k_edge_sum1<<<gr(Et, T), T, 0, stream>>>(ei, asc1, adt1, emax1, den1, E, N);
    k_edge_msg1<<<gr((long)Et * 64, T), T, 0, stream>>>(ei, asc1, adt1, emax1, den1,
                                                        xh1, out1, E, N);
    k_bias_elu1<<<gr((long)N * F1, T), T, 0, stream>>>(out1, b1, hbf, (long)N * F1);

    // ---- layer 2 ----
    k_zero<<<gr((long)N * HID, T), T, 0, stream>>>((unsigned*)out2, (long)N * HID);
    k_zero<<<gr(N, T), T, 0, stream>>>(emax2, N);
    k_zero<<<gr(N, T), T, 0, stream>>>((unsigned*)den2, N);
    k_zero<<<1, T, 0, stream>>>((unsigned*)pool, G * HID);
    k_zero<<<1, G, 0, stream>>>((unsigned*)cnts, G);

    int mtiles = (N + 15) / 16;
    k_gemm2<<<(mtiles + 1) / 2, 256, 0, stream>>>(hbf, w2bf, xh2, N);
    k_attn2<<<gr((long)N * 32, T), T, 0, stream>>>(xh2, as2, ad2, asc2, adt2, N);

    k_edge_max2<<<gr(Et, T), T, 0, stream>>>(ei, asc2, adt2, emax2, E, N);
    k_edge_sum2<<<gr(Et, T), T, 0, stream>>>(ei, asc2, adt2, emax2, den2, E, N);
    k_edge_msg2<<<gr((long)Et * 16, T), T, 0, stream>>>(ei, asc2, adt2, emax2, den2,
                                                        xh2, out2, E, N);

    // ---- pool + head ----
    k_pool<<<gr((long)N * HID, T), T, 0, stream>>>(out2, b2, bat, pool, cnts, N);
    k_final<<<1, 128, 0, stream>>>(pool, cnts, Wl, bl, (float*)d_out);
}
